// SwitchDropTokenCalculator_63694364999794
// MI455X (gfx1250) — matrix operand order repaired and checked by reading, not compiled, not run on hardware
//
#include <hip/hip_runtime.h>

// ---------------------------------------------------------------------------
// Switch-MoE (top-1, capacity drop) with SwiGLU expert FFN for MI455X/gfx1250.
// Grouped GEMMs on v_wmma_f32_16x16x32_bf16 (fp32 weights converted to bf16
// on the fly, fp32 accumulation), software-pipelined double-buffered LDS,
// A-tiles streamed with global_load_async_to_lds_b128 (ASYNCcnt).
// N=8192, d=1024, E=8, h=2816, C=1280.
// ---------------------------------------------------------------------------

#define MOE_N 8192
#define MOE_D 1024
#define MOE_E 8
#define MOE_H 2816
#define MOE_C 1280   // int(1.25 * N / E)

typedef __attribute__((ext_vector_type(16))) __bf16 v16bf;
typedef __attribute__((ext_vector_type(8)))  __bf16 v8bf;
typedef __attribute__((ext_vector_type(2)))  __bf16 v2bf;
typedef __attribute__((ext_vector_type(8)))  float  v8f;

union V16U { v16bf v; v8bf h[2]; };

__device__ __forceinline__ unsigned short f32_to_bf16(float f) {
    union { float f; unsigned u; } x; x.f = f;
    unsigned r = x.u + 0x7FFFu + ((x.u >> 16) & 1u);   // round-to-nearest-even
    return (unsigned short)(r >> 16);
}

// Pack two floats to packed bf16x2 (one v_cvt_pk_bf16_f32 when available).
__device__ __forceinline__ unsigned pack2bf(float a, float b) {
#if __has_builtin(__builtin_amdgcn_cvt_pk_bf16_f32)
    union { v2bf v; unsigned u; } r;
    r.v = __builtin_amdgcn_cvt_pk_bf16_f32(a, b);
    return r.u;
#else
    return (unsigned)f32_to_bf16(a) | ((unsigned)f32_to_bf16(b) << 16);
#endif
}

__device__ __forceinline__ v8f vzero8() {
    v8f v;
#pragma unroll
    for (int i = 0; i < 8; ++i) v[i] = 0.0f;
    return v;
}

// Async global -> LDS copy, 16B per lane, tracked on ASYNCcnt (CDNA5).
__device__ __forceinline__ void async_ld_b128(unsigned lds_byte_addr,
                                              const unsigned short* gptr) {
    asm volatile("global_load_async_to_lds_b128 %0, %1, off"
                 :: "v"(lds_byte_addr), "v"(gptr) : "memory");
}
__device__ __forceinline__ void wait_async0() {
    asm volatile("s_wait_asynccnt 0x0" ::: "memory");
}
// Low 32 bits of a flat pointer into the LDS aperture == LDS byte address.
__device__ __forceinline__ unsigned lds_addr(const void* p) {
    return (unsigned)(size_t)p;
}

// ---------------------------------------------------------------------------
// 1) Routing: FCFS position per token within its expert queue (capacity drop).
// ---------------------------------------------------------------------------
__global__ __launch_bounds__(256) void route_kernel(const long long* __restrict__ topk_idx,
                                                    int* __restrict__ slot_token,
                                                    int* __restrict__ token_slot) {
    __shared__ int scnt[256 * MOE_E];
    const int tid = threadIdx.x;

    for (int s = tid; s < MOE_E * MOE_C; s += 256) slot_token[s] = -1;

    const int chunk = MOE_N / 256;          // 32 tokens per thread
    const int base  = tid * chunk;
    int cnt[MOE_E];
#pragma unroll
    for (int e = 0; e < MOE_E; ++e) cnt[e] = 0;
    for (int i = 0; i < chunk; ++i) cnt[(int)topk_idx[base + i]]++;
#pragma unroll
    for (int e = 0; e < MOE_E; ++e) scnt[tid * MOE_E + e] = cnt[e];
    __syncthreads();

    if (tid < MOE_E) {                      // 8-way parallel exclusive scan
        int run = 0;
        for (int t = 0; t < 256; ++t) {
            int v = scnt[t * MOE_E + tid];
            scnt[t * MOE_E + tid] = run;
            run += v;
        }
    }
    __syncthreads();

    int loc[MOE_E];
#pragma unroll
    for (int e = 0; e < MOE_E; ++e) loc[e] = 0;
    for (int i = 0; i < chunk; ++i) {
        int tok = base + i;
        int e   = (int)topk_idx[tok];
        int pos = scnt[tid * MOE_E + e] + loc[e]++;
        if (pos < MOE_C) {
            int s = e * MOE_C + pos;
            token_slot[tok] = s;
            slot_token[s]   = tok;
        } else {
            token_slot[tok] = -1;
        }
    }
}

// ---------------------------------------------------------------------------
// 2) Dispatch: gather token rows into bf16 xe[E*C, d] (zeros for empty slots).
// ---------------------------------------------------------------------------
__global__ __launch_bounds__(256) void dispatch_kernel(const float* __restrict__ x,
                                                       const int* __restrict__ slot_token,
                                                       unsigned short* __restrict__ xe) {
    const int s = blockIdx.x;
    const int t = slot_token[s];
    const int c = threadIdx.x * 4;
    float4 v = make_float4(0.f, 0.f, 0.f, 0.f);
    if (t >= 0) v = *(const float4*)(x + (size_t)t * MOE_D + c);
    uint2 p;
    p.x = pack2bf(v.x, v.y);
    p.y = pack2bf(v.z, v.w);
    *(uint2*)(xe + (size_t)s * MOE_D + c) = p;
}

// ---------------------------------------------------------------------------
// 3) Pass-through for dropped / unrouted tokens: y = x.
// ---------------------------------------------------------------------------
__global__ __launch_bounds__(256) void passthrough_kernel(const float* __restrict__ x,
                                                          const int* __restrict__ token_slot,
                                                          float* __restrict__ y) {
    const int t = blockIdx.x;
    if (token_slot[t] >= 0) return;
    const int c = threadIdx.x * 4;
    *(float4*)(y + (size_t)t * MOE_D + c) = *(const float4*)(x + (size_t)t * MOE_D + c);
}

// ---------------------------------------------------------------------------
// 4) GEMM1: hact = silu(xe @ Wg[e]) * (xe @ Wu[e])      [C x d] * [d x h]
//    128x128 tile / block, 8 wave32 waves (2x4), wave tile 64x32 (4x2 WMMA),
//    double-buffered LDS, async A copies, pipelined B convert.
// ---------------------------------------------------------------------------
__global__ __launch_bounds__(256, 1) void gemm1_kernel(const unsigned short* __restrict__ xe,
                                                       const float* __restrict__ Wg,
                                                       const float* __restrict__ Wu,
                                                       unsigned short* __restrict__ hact) {
    __shared__ __align__(32) unsigned short sA [2][128 * 32];
    __shared__ __align__(32) unsigned short sBg[2][32 * 128];
    __shared__ __align__(32) unsigned short sBu[2][32 * 128];

    const int e  = blockIdx.z;
    const int m0 = blockIdx.y * 128;
    const int n0 = blockIdx.x * 128;
    const int tid  = threadIdx.x;
    const int lane = tid & 31;
    const int w    = tid >> 5;
    const int wm   = w >> 2;        // 0..1
    const int wn   = w & 3;         // 0..3

    const int aRow = tid >> 1;             // 0..127
    const int aCol = (tid & 1) * 16;       // 0 / 16
    const int bRow = tid >> 3;             // 0..31
    const int bCol = (tid & 7) * 16;       // 0..112

    const unsigned short* Aptr = xe + (size_t)(e * MOE_C + m0 + aRow) * MOE_D + aCol;
    const float* Bg = Wg + (size_t)e * MOE_D * MOE_H + (size_t)bRow * MOE_H + n0 + bCol;
    const float* Bu = Wu + (size_t)e * MOE_D * MOE_H + (size_t)bRow * MOE_H + n0 + bCol;

    unsigned aLds[2];
#pragma unroll
    for (int b = 0; b < 2; ++b)
        aLds[b] = lds_addr(&sA[b][aRow * 32 + aCol]);

    v8f accG[4][2], accU[4][2];
#pragma unroll
    for (int i = 0; i < 4; ++i)
#pragma unroll
        for (int j = 0; j < 2; ++j) { accG[i][j] = vzero8(); accU[i][j] = vzero8(); }

    float4 rg[4], ru[4];
    const int NK = MOE_D / 32;             // 32 K-steps

    // ---- prologue: stage tile 0 into buffer 0 ----
    async_ld_b128(aLds[0],      Aptr);
    async_ld_b128(aLds[0] + 16, Aptr + 8);
#pragma unroll
    for (int q = 0; q < 4; ++q) {
        rg[q] = *(const float4*)(Bg + q * 4);
        ru[q] = *(const float4*)(Bu + q * 4);
    }
#pragma unroll
    for (int q = 0; q < 4; ++q) {
        uint2 pg, pu;
        pg.x = pack2bf(rg[q].x, rg[q].y); pg.y = pack2bf(rg[q].z, rg[q].w);
        pu.x = pack2bf(ru[q].x, ru[q].y); pu.y = pack2bf(ru[q].z, ru[q].w);
        *(uint2*)(sBg[0] + bRow * 128 + bCol + q * 4) = pg;
        *(uint2*)(sBu[0] + bRow * 128 + bCol + q * 4) = pu;
    }
    wait_async0();
    __syncthreads();

    for (int k = 0; k < NK; ++k) {
        const int buf = k & 1;
        const bool hasNext = (k + 1 < NK);

        if (hasNext) {
            const int kk2 = (k + 1) * 32;
            // async A copy for k+1 into the other buffer (readers of it
            // finished at the end-of-(k-1) barrier)
            async_ld_b128(aLds[1 - buf],      Aptr + kk2);
            async_ld_b128(aLds[1 - buf] + 16, Aptr + kk2 + 8);
            // issue B global loads for k+1 (land in regs during compute)
#pragma unroll
            for (int q = 0; q < 4; ++q) {
                rg[q] = *(const float4*)(Bg + (size_t)kk2 * MOE_H + q * 4);
                ru[q] = *(const float4*)(Bu + (size_t)kk2 * MOE_H + q * 4);
            }
            if (k + 2 < NK) {   // L2 prefetch two tiles ahead (global_prefetch_b8)
                __builtin_prefetch(Bg + (size_t)(k + 2) * 32 * MOE_H, 0, 0);
                __builtin_prefetch(Bu + (size_t)(k + 2) * 32 * MOE_H, 0, 0);
            }
        }

        // ---- compute from lds[buf] ----
        const int ar  = lane & 15;
        const int akb = (lane >> 4) * 8;
        v16bf afr[4];
#pragma unroll
        for (int mt = 0; mt < 4; ++mt) {
            const int row = wm * 64 + mt * 16 + ar;
            V16U t;
            t.h[0] = *(const v8bf*)(sA[buf] + row * 32 + akb);
            t.h[1] = *(const v8bf*)(sA[buf] + row * 32 + akb + 16);
            afr[mt] = t.v;
        }
        v16bf bgf[2], buf2[2];
#pragma unroll
        for (int nt = 0; nt < 2; ++nt) {
            const int col = wn * 32 + nt * 16;
#pragma unroll
            for (int s = 0; s < 16; ++s) {
                const int kr = (s >> 3) * 16 + (lane >> 4) * 8 + (s & 7);
                bgf[nt][s]  = *(const __bf16*)(sBg[buf] + kr * 128 + col + (lane & 15));
                buf2[nt][s] = *(const __bf16*)(sBu[buf] + kr * 128 + col + (lane & 15));
            }
        }
#pragma unroll
        for (int mt = 0; mt < 4; ++mt)
#pragma unroll
            for (int nt = 0; nt < 2; ++nt) {
                accG[mt][nt] = __builtin_amdgcn_wmma_f32_16x16x32_bf16(
                    false, afr[mt], false, bgf[nt], (short)0, accG[mt][nt], false, false);
                accU[mt][nt] = __builtin_amdgcn_wmma_f32_16x16x32_bf16(
                    false, afr[mt], false, buf2[nt], (short)0, accU[mt][nt], false, false);
            }

        if (hasNext) {
            // convert + store B tile k+1 into the other buffer
#pragma unroll
            for (int q = 0; q < 4; ++q) {
                uint2 pg, pu;
                pg.x = pack2bf(rg[q].x, rg[q].y); pg.y = pack2bf(rg[q].z, rg[q].w);
                pu.x = pack2bf(ru[q].x, ru[q].y); pu.y = pack2bf(ru[q].z, ru[q].w);
                *(uint2*)(sBg[1 - buf] + bRow * 128 + bCol + q * 4) = pg;
                *(uint2*)(sBu[1 - buf] + bRow * 128 + bCol + q * 4) = pu;
            }
            wait_async0();
            __syncthreads();
        }
    }

    // ---- fused SwiGLU epilogue, bf16 store ----
#pragma unroll
    for (int mt = 0; mt < 4; ++mt)
#pragma unroll
        for (int nt = 0; nt < 2; ++nt)
#pragma unroll
            for (int j = 0; j < 8; ++j) {
                const int m = m0 + wm * 64 + mt * 16 + j + 8 * (lane >> 4);
                const int n = n0 + wn * 32 + nt * 16 + (lane & 15);
                const float g = accG[mt][nt][j];
                const float u = accU[mt][nt][j];
                const float a = (g / (1.0f + __expf(-g))) * u;   // silu(g)*u
                hact[(size_t)(e * MOE_C + m) * MOE_H + n] = f32_to_bf16(a);
            }
}

// ---------------------------------------------------------------------------
// 5) GEMM2: out = hact @ Wd[e]  then scatter y[token] = out * score[token].
// ---------------------------------------------------------------------------
__global__ __launch_bounds__(256, 1) void gemm2_kernel(const unsigned short* __restrict__ hact,
                                                       const float* __restrict__ Wd,
                                                       const int* __restrict__ slot_token,
                                                       const float* __restrict__ scores,
                                                       float* __restrict__ y) {
    __shared__ __align__(32) unsigned short sA[2][128 * 32];
    __shared__ __align__(32) unsigned short sB[2][32 * 128];

    const int e  = blockIdx.z;
    const int m0 = blockIdx.y * 128;
    const int n0 = blockIdx.x * 128;
    const int tid  = threadIdx.x;
    const int lane = tid & 31;
    const int w    = tid >> 5;
    const int wm   = w >> 2;
    const int wn   = w & 3;

    const int aRow = tid >> 1;
    const int aCol = (tid & 1) * 16;
    const int bRow = tid >> 3;
    const int bCol = (tid & 7) * 16;

    const unsigned short* Aptr = hact + (size_t)(e * MOE_C + m0 + aRow) * MOE_H + aCol;
    const float* Bd = Wd + (size_t)e * MOE_H * MOE_D + (size_t)bRow * MOE_D + n0 + bCol;

    unsigned aLds[2];
#pragma unroll
    for (int b = 0; b < 2; ++b)
        aLds[b] = lds_addr(&sA[b][aRow * 32 + aCol]);

    v8f acc[4][2];
#pragma unroll
    for (int i = 0; i < 4; ++i)
#pragma unroll
        for (int j = 0; j < 2; ++j) acc[i][j] = vzero8();

    float4 rb[4];
    const int NK = MOE_H / 32;             // 88 K-steps

    // ---- prologue ----
    async_ld_b128(aLds[0],      Aptr);
    async_ld_b128(aLds[0] + 16, Aptr + 8);
#pragma unroll
    for (int q = 0; q < 4; ++q) rb[q] = *(const float4*)(Bd + q * 4);
#pragma unroll
    for (int q = 0; q < 4; ++q) {
        uint2 p;
        p.x = pack2bf(rb[q].x, rb[q].y); p.y = pack2bf(rb[q].z, rb[q].w);
        *(uint2*)(sB[0] + bRow * 128 + bCol + q * 4) = p;
    }
    wait_async0();
    __syncthreads();

    for (int k = 0; k < NK; ++k) {
        const int buf = k & 1;
        const bool hasNext = (k + 1 < NK);

        if (hasNext) {
            const int kk2 = (k + 1) * 32;
            async_ld_b128(aLds[1 - buf],      Aptr + kk2);
            async_ld_b128(aLds[1 - buf] + 16, Aptr + kk2 + 8);
#pragma unroll
            for (int q = 0; q < 4; ++q)
                rb[q] = *(const float4*)(Bd + (size_t)kk2 * MOE_D + q * 4);
            if (k + 2 < NK)
                __builtin_prefetch(Bd + (size_t)(k + 2) * 32 * MOE_D, 0, 0);
        }

        const int ar  = lane & 15;
        const int akb = (lane >> 4) * 8;
        v16bf afr[4];
#pragma unroll
        for (int mt = 0; mt < 4; ++mt) {
            const int row = wm * 64 + mt * 16 + ar;
            V16U t;
            t.h[0] = *(const v8bf*)(sA[buf] + row * 32 + akb);
            t.h[1] = *(const v8bf*)(sA[buf] + row * 32 + akb + 16);
            afr[mt] = t.v;
        }
        v16bf bfr[2];
#pragma unroll
        for (int nt = 0; nt < 2; ++nt)
#pragma unroll
            for (int s = 0; s < 16; ++s) {
                const int kr = (s >> 3) * 16 + (lane >> 4) * 8 + (s & 7);
                bfr[nt][s] = *(const __bf16*)(sB[buf] + kr * 128 + wn * 32 + nt * 16 + (lane & 15));
            }
#pragma unroll
        for (int mt = 0; mt < 4; ++mt)
#pragma unroll
            for (int nt = 0; nt < 2; ++nt)
                acc[mt][nt] = __builtin_amdgcn_wmma_f32_16x16x32_bf16(
                    false, afr[mt], false, bfr[nt], (short)0, acc[mt][nt], false, false);

        if (hasNext) {
#pragma unroll
            for (int q = 0; q < 4; ++q) {
                uint2 p;
                p.x = pack2bf(rb[q].x, rb[q].y); p.y = pack2bf(rb[q].z, rb[q].w);
                *(uint2*)(sB[1 - buf] + bRow * 128 + bCol + q * 4) = p;
            }
            wait_async0();
            __syncthreads();
        }
    }

    // ---- scatter epilogue: y[token] = out * score ----
#pragma unroll
    for (int mt = 0; mt < 4; ++mt) {
#pragma unroll
        for (int j = 0; j < 8; ++j) {
            const int mrow = m0 + wm * 64 + mt * 16 + j + 8 * (lane >> 4);
            const int tok  = slot_token[e * MOE_C + mrow];
            if (tok < 0) continue;
            const float sc = scores[tok];
#pragma unroll
            for (int nt = 0; nt < 2; ++nt) {
                const int n = n0 + wn * 32 + nt * 16 + (lane & 15);
                y[(size_t)tok * MOE_D + n] = acc[mt][nt][j] * sc;
            }
        }
    }
}

// ---------------------------------------------------------------------------
// Host launcher
// ---------------------------------------------------------------------------
extern "C" void kernel_launch(void* const* d_in, const int* in_sizes, int n_in,
                              void* d_out, int out_size, void* d_ws, size_t ws_size,
                              hipStream_t stream) {
    const float*     x      = (const float*)d_in[0];
    const long long* tk_idx = (const long long*)d_in[1];   // int64 expert ids
    const float*     scores = (const float*)d_in[2];
    const float*     Wg     = (const float*)d_in[3];
    const float*     Wu     = (const float*)d_in[4];
    const float*     Wd     = (const float*)d_in[5];
    float*           y      = (float*)d_out;

    // workspace layout
    char* ws = (char*)d_ws;
    int* slot_token = (int*)ws;                               //  E*C ints   (40 KB)
    int* token_slot = (int*)(ws + 40960);                     //  N ints     (32 KB)
    unsigned short* xe   = (unsigned short*)(ws + 73728);                 // 21 MB bf16
    unsigned short* hact = (unsigned short*)(ws + 73728 + 20971520);      // 58 MB bf16

    route_kernel<<<1, 256, 0, stream>>>(tk_idx, slot_token, token_slot);
    dispatch_kernel<<<MOE_E * MOE_C, 256, 0, stream>>>(x, slot_token, xe);
    passthrough_kernel<<<MOE_N, 256, 0, stream>>>(x, token_slot, y);
    gemm1_kernel<<<dim3(MOE_H / 128, MOE_C / 128, MOE_E), 256, 0, stream>>>(xe, Wg, Wu, hact);
    gemm2_kernel<<<dim3(MOE_D / 128, MOE_C / 128, MOE_E), 256, 0, stream>>>(hact, Wd, slot_token, scores, y);
}
